// DimeNetPP_63531156242867
// MI455X (gfx1250) — matrix operand order repaired and checked by reading, not compiled, not run on hardware
//
#include <hip/hip_runtime.h>
#include <hip/hip_bf16.h>

typedef __attribute__((ext_vector_type(16))) __bf16 bf16x16;
typedef __attribute__((ext_vector_type(8)))  __bf16 bf16x8;
typedef __attribute__((ext_vector_type(8)))  float  f32x8;
typedef __attribute__((ext_vector_type(4)))  unsigned int u32x4;
typedef __attribute__((ext_vector_type(8)))  int i32x8;
typedef __attribute__((ext_vector_type(4)))  int i32x4;

__device__ __forceinline__ f32x8 wmma_bf16(bf16x16 a, bf16x16 b, f32x8 c) {
  // (neg_a, A, neg_b, B, c_mod, C, reuse_a, reuse_b)
  return __builtin_amdgcn_wmma_f32_16x16x32_bf16(false, a, false, b, (short)0, c, false, false);
}

__device__ __forceinline__ float silu_f(float x) {
  return x * (1.0f / (1.0f + __expf(-x)));
}

// ---- Tensor Data Mover: 1-D contiguous copy global->LDS (bytes % 8 == 0). ----
// D# per CDNA5 ISA 8.3-8.6: group0 {count=1, lds_addr, global_addr(57b), type=2},
// group1 {data_size=8B, tensor_dim0=n, tensor_dim1=1, tile_dim0=n, tile_dim1=1,
// strides=n}. Groups 2/3 (and the trailing group word block) zero: dims 2..4 unused.
// Issued by one wave; EXEC ignored. clang-23 toolchain: 6-arg builtin form.
__device__ __forceinline__ void tdm_load_1d(const void* gsrc, void* lds_dst, unsigned bytes) {
  const unsigned n = bytes >> 3;                         // 8-byte elements
  const unsigned long long ga = (unsigned long long)(size_t)gsrc;
  const unsigned lds = (unsigned)(unsigned long long)(size_t)lds_dst; // low 32 = LDS offset
  u32x4 g0;
  g0[0] = 1u;                                            // count=1 (valid user D#)
  g0[1] = lds;                                           // lds_addr
  g0[2] = (unsigned)(ga & 0xFFFFFFFFu);                  // global_addr[31:0]
  g0[3] = (unsigned)((ga >> 32) & 0x01FFFFFFu) | (2u << 30); // global_addr[56:32], type=2
  i32x8 g1;
  g1[0] = (int)(3u << 16);                               // wg_mask=0, data_size=3 (8B)
  g1[1] = (int)((n & 0xFFFFu) << 16);                    // tensor_dim0[15:0]
  g1[2] = (int)(((n >> 16) & 0xFFFFu) | (1u << 16));     // tensor_dim0[31:16], tensor_dim1=1
  g1[3] = (int)((n & 0xFFFFu) << 16);                    // tensor_dim1 hi=0, tile_dim0=n
  g1[4] = 1;                                             // tile_dim1=1, tile_dim2=0
  g1[5] = (int)n;                                        // tensor_dim0_stride[31:0]
  g1[6] = (int)((n & 0xFFFFu) << 16);                    // d0_stride hi=0, d1_stride[15:0]
  g1[7] = (int)((n >> 16) & 0xFFFFu);                    // d1_stride[47:16]
  i32x4 z4; z4[0] = 0; z4[1] = 0; z4[2] = 0; z4[3] = 0;
  i32x8 z8;
#pragma unroll
  for (int i = 0; i < 8; ++i) z8[i] = 0;
  __builtin_amdgcn_tensor_load_to_lds(g0, g1, z4, z4, z8, 0);
}

#define TDM_ISSUE(SRC, DST, BYTES) do { if (tid < 32) tdm_load_1d((SRC), (DST), (BYTES)); } while (0)
#define TDM_WAIT1()                do { if (tid < 32) __builtin_amdgcn_s_wait_tensorcnt(1); } while (0)
#define TDM_WAIT0()                do { if (tid < 32) __builtin_amdgcn_s_wait_tensorcnt(0); } while (0)

// A fragment: 16x32 bf16 tile, rows row_base..row_base+15, K = k_base..k_base+31.
// Order used here (same as B): lanes 0-15 hold M=lane, K 0..15; lanes 16-31 hold M=lane-16, K 16..31.
__device__ __forceinline__ bf16x16 load_a_frag(const __bf16* tile, int ldk,
                                               int row_base, int k_base, int lane) {
  const int r = row_base + (lane & 15);
  const int k_off = (lane >> 4) << 4;          // 0 or 16
  const __bf16* p = tile + r * ldk + k_base + k_off;
  bf16x8 lo = *(const bf16x8*)(p);             // K = k_off .. k_off+7
  bf16x8 hi = *(const bf16x8*)(p + 8);         // K = k_off+8 .. k_off+15
  bf16x16 out;
#pragma unroll
  for (int i = 0; i < 8; ++i) { out[i] = lo[i]; out[i + 8] = hi[i]; }
  return out;
}

// B fragment: 32x16 bf16, from W^T stored [N][K] row-major (LDS or global).
// lanes 0-15: N=lane, K 0..15; lanes 16-31: N=lane-16, K 16..31.
__device__ __forceinline__ bf16x16 load_b_frag(const __bf16* WT, int ldk,
                                               int n_base, int k_base, int lane) {
  const int n = n_base + (lane & 15);
  const int k_off = (lane >> 4) << 4;          // 0 or 16
  return *(const bf16x16*)(WT + n * ldk + k_base + k_off);
}

// One wave computes a 16 x (NT*16) output tile: A rows + B = W^T [NT*16][KTOT], both in LDS.
template<int KTOT, int NT>
__device__ __forceinline__ void wave_gemm(const __bf16* a_tile, int lda,
                                          const __bf16* WT,
                                          int row_base, int lane, f32x8* acc) {
#pragma unroll
  for (int nt = 0; nt < NT; ++nt)
#pragma unroll
    for (int i = 0; i < 8; ++i) acc[nt][i] = 0.0f;
#pragma unroll
  for (int kb = 0; kb < KTOT; kb += 32) {
    bf16x16 a = load_a_frag(a_tile, lda, row_base, kb, lane);
#pragma unroll
    for (int nt = 0; nt < NT; ++nt) {
      bf16x16 b = load_b_frag(WT, KTOT, nt * 16, kb, lane);
      acc[nt] = wmma_bf16(a, b, acc[nt]);
    }
  }
}

// ---------------- weight conversion: W[k][n] fp32 -> WT[n][k] bf16 ----------------
__global__ void k_w_to_bf16_t(const float* __restrict__ W, __bf16* __restrict__ WT, int K, int N) {
  int idx = blockIdx.x * 256 + threadIdx.x;
  if (idx >= K * N) return;
  int k = idx / N, n = idx - k * N;
  WT[n * K + k] = (__bf16)W[k * N + n];
}

// ---------------- stage 1: m_ang = silu( (silu(x@Wkj+bkj) * rbf-mod) @ Wdown ) ----------------
__global__ __launch_bounds__(256)
void k_edge_pre(const float* __restrict__ m_input, const float* __restrict__ rbf,
                const float* __restrict__ Wrbf1, const float* __restrict__ Wrbf2,
                const float* __restrict__ bkj,
                const __bf16* __restrict__ WkjT, const __bf16* __restrict__ WdownT,
                float* __restrict__ m_ang, int E) {
  extern __shared__ char smem[];
  __bf16* tileX = (__bf16*)smem;                 // 32 KB
  __bf16* tileY = (__bf16*)(smem + 32768);       // 32 KB
  float*  h8    = (float*)(smem + 65536);        // 4 KB
  __bf16* wKj   = (__bf16*)(smem + 69632);       // 32 KB  (TDM-staged Wkj^T)
  __bf16* wDn   = (__bf16*)(smem + 102400);      // 16 KB  (TDM-staged Wdown^T)
  const int tid = threadIdx.x;
  const int e0 = blockIdx.x * 128;

  TDM_ISSUE(WkjT, wKj, 128 * 128 * 2);
  TDM_ISSUE(WdownT, wDn, 64 * 128 * 2);

  if (tid == 0 && e0 + 128 < E)
    __builtin_prefetch(m_input + (size_t)(e0 + 128) * 128, 0, 1);

  for (int idx = tid; idx < 128 * 128; idx += 256) {
    int r = idx >> 7, c = idx & 127;
    float v = (e0 + r < E) ? m_input[(size_t)(e0 + r) * 128 + c] : 0.0f;
    tileX[idx] = (__bf16)v;
  }
  for (int idx = tid; idx < 128 * 8; idx += 256) {
    int r = idx >> 3, j = idx & 7;
    float s = 0.0f;
    if (e0 + r < E) {
#pragma unroll
      for (int i = 0; i < 6; ++i) s += rbf[(size_t)(e0 + r) * 6 + i] * Wrbf1[i * 8 + j];
    }
    h8[idx] = s;
  }
  TDM_WAIT0();
  __syncthreads();

  const int lane = tid & 31;
  const int row_base = (tid >> 5) * 16;
  const int n15 = lane & 15;
  const int m_off = (lane >> 4) * 8;

  f32x8 acc[8];
  wave_gemm<128, 8>(tileX, 128, wKj, row_base, lane, acc);
#pragma unroll
  for (int nt = 0; nt < 8; ++nt) {
    int n = nt * 16 + n15;
    float b = bkj[n];
#pragma unroll
    for (int v = 0; v < 8; ++v) {
      int m = row_base + m_off + v;
      float x = silu_f(acc[nt][v] + b);
      float rm = 0.0f;
#pragma unroll
      for (int j = 0; j < 8; ++j) rm += h8[m * 8 + j] * Wrbf2[j * 128 + n];
      tileY[m * 128 + n] = (__bf16)(x * rm);
    }
  }
  __syncthreads();

  f32x8 acc2[4];
  wave_gemm<128, 4>(tileY, 128, wDn, row_base, lane, acc2);
#pragma unroll
  for (int nt = 0; nt < 4; ++nt) {
    int n = nt * 16 + n15;
#pragma unroll
    for (int v = 0; v < 8; ++v) {
      int m = row_base + m_off + v;
      int e = e0 + m;
      if (e < E) m_ang[(size_t)e * 64 + n] = silu_f(acc2[nt][v]);
    }
  }
}

// ---------------- stage 2: one wave32 per triplet; gather + SBF mod + scatter-add ----------------
__global__ __launch_bounds__(256)
void k_triplet(const float* __restrict__ sbf, const float* __restrict__ Wsbf1,
               const float* __restrict__ Wsbf2, const float* __restrict__ m_ang,
               const int* __restrict__ expand_to_kj, const int* __restrict__ reduce_to_ji,
               float* __restrict__ agg, int T) {
  const int lane = threadIdx.x & 31;
  const int t = blockIdx.x * 8 + (threadIdx.x >> 5);
  if (t >= T) return;

  float hj = 0.0f;                       // lanes 0..7 each compute one column of sbf@Wsbf1
  if (lane < 8) {
    const float* s = sbf + (size_t)t * 42;
#pragma unroll 6
    for (int i = 0; i < 42; ++i) hj += s[i] * Wsbf1[i * 8 + lane];
  }
  float h[8];
#pragma unroll
  for (int j = 0; j < 8; ++j) h[j] = __shfl(hj, j, 32);

  const int e_src = expand_to_kj[t];
  const int e_dst = reduce_to_ji[t];
#pragma unroll
  for (int half = 0; half < 2; ++half) {
    int d = lane + half * 32;
    float o = 0.0f;
#pragma unroll
    for (int j = 0; j < 8; ++j) o += h[j] * Wsbf2[j * 64 + d];
    float val = m_ang[(size_t)e_src * 64 + d] * o;   // gather (L2-resident table)
    atomicAdd(&agg[(size_t)e_dst * 64 + d], val);    // global_atomic_add_f32
  }
}

// ---------------- stage 3: fused edge tail; weights double-buffered via TDM ----------------
__global__ __launch_bounds__(256)
void k_edge_post(const float* __restrict__ m_input, const float* __restrict__ agg,
                 const __bf16* __restrict__ WupT, const __bf16* __restrict__ WjiT,
                 const float* __restrict__ bji,
                 const __bf16* __restrict__ Wb1aT, const float* __restrict__ bb1a,
                 const __bf16* __restrict__ Wb1bT, const float* __restrict__ bb1b,
                 const __bf16* __restrict__ WfinT, const float* __restrict__ bfin,
                 const __bf16* __restrict__ Wa1aT, const float* __restrict__ ba1a,
                 const __bf16* __restrict__ Wa1bT, const float* __restrict__ ba1b,
                 const __bf16* __restrict__ Wa2aT, const float* __restrict__ ba2a,
                 const __bf16* __restrict__ Wa2bT, const float* __restrict__ ba2b,
                 float* __restrict__ out, int E) {
  extern __shared__ char smem[];
  __bf16* tileA = (__bf16*)smem;             // 32 KB
  __bf16* tileB = (__bf16*)(smem + 32768);   // 32 KB
  __bf16* wbuf0 = (__bf16*)(smem + 65536);   // 32 KB  TDM weight buffer 0
  __bf16* wbuf1 = (__bf16*)(smem + 98304);   // 32 KB  TDM weight buffer 1
  const int tid = threadIdx.x;
  const int e0 = blockIdx.x * 128;
  const int lane = tid & 31;
  const int row_base = (tid >> 5) * 16;
  const int n15 = lane & 15;
  const int m_off = (lane >> 4) * 8;

  // Prime the TDM pipeline: Wup -> wbuf0, Wji -> wbuf1.
  TDM_ISSUE(WupT, wbuf0, 128 * 64 * 2);
  TDM_ISSUE(WjiT, wbuf1, 128 * 128 * 2);

  if (tid == 0 && e0 + 128 < E)
    __builtin_prefetch(agg + (size_t)(e0 + 128) * 64, 0, 1);

  // agg tile -> tileA ([128][64] bf16); m_input tile -> tileB ([128][128] bf16)
  for (int idx = tid; idx < 128 * 64; idx += 256) {
    int r = idx >> 6, c = idx & 63;
    float v = (e0 + r < E) ? agg[(size_t)(e0 + r) * 64 + c] : 0.0f;
    tileA[idx] = (__bf16)v;
  }
  for (int idx = tid; idx < 128 * 128; idx += 256) {
    int r = idx >> 7, c = idx & 127;
    float v = (e0 + r < E) ? m_input[(size_t)(e0 + r) * 128 + c] : 0.0f;
    tileB[idx] = (__bf16)v;
  }
  TDM_WAIT1();            // Wup landed (Wji may still be in flight)
  __syncthreads();

  // stage 0: prop = silu(agg @ Wup)  (K=64), from wbuf0
  f32x8 prop[8];
  wave_gemm<64, 8>(tileA, 64, wbuf0, row_base, lane, prop);
#pragma unroll
  for (int nt = 0; nt < 8; ++nt)
#pragma unroll
    for (int v = 0; v < 8; ++v) prop[nt][v] = silu_f(prop[nt][v]);
  __syncthreads();                         // wbuf0 + agg region of tileA drained
  TDM_ISSUE(Wb1aT, wbuf0, 32768);
  TDM_WAIT1();                             // Wji ready
  __syncthreads();

  f32x8 acc[8];
  // stage 1: t0 = silu(m_input @ Wji + bji) + prop -> tileA   (B = wbuf1)
  wave_gemm<128, 8>(tileB, 128, wbuf1, row_base, lane, acc);
#pragma unroll
  for (int nt = 0; nt < 8; ++nt) {
    int n = nt * 16 + n15; float b = bji[n];
#pragma unroll
    for (int v = 0; v < 8; ++v) {
      int m = row_base + m_off + v;
      tileA[m * 128 + n] = (__bf16)(silu_f(acc[nt][v] + b) + prop[nt][v]);
    }
  }
  __syncthreads();
  TDM_ISSUE(Wb1bT, wbuf1, 32768);
  TDM_WAIT1();                             // Wb1a ready
  __syncthreads();

  // stage 2: resb1 p1: tileB = silu(t0 @ W1 + b1)   (B = wbuf0)
  wave_gemm<128, 8>(tileA, 128, wbuf0, row_base, lane, acc);
#pragma unroll
  for (int nt = 0; nt < 8; ++nt) {
    int n = nt * 16 + n15; float b = bb1a[n];
#pragma unroll
    for (int v = 0; v < 8; ++v) {
      int m = row_base + m_off + v;
      tileB[m * 128 + n] = (__bf16)silu_f(acc[nt][v] + b);
    }
  }
  __syncthreads();
  TDM_ISSUE(WfinT, wbuf0, 32768);
  TDM_WAIT1();                             // Wb1b ready
  __syncthreads();

  // stage 3: resb1 p2: t1 = t0 + silu(h @ W2 + b2) -> tileB   (B = wbuf1)
  wave_gemm<128, 8>(tileB, 128, wbuf1, row_base, lane, acc);
#pragma unroll
  for (int nt = 0; nt < 8; ++nt) {
    int n = nt * 16 + n15; float b = bb1b[n];
#pragma unroll
    for (int v = 0; v < 8; ++v) {
      int m = row_base + m_off + v;
      tileB[m * 128 + n] = (__bf16)((float)tileA[m * 128 + n] + silu_f(acc[nt][v] + b));
    }
  }
  __syncthreads();
  TDM_ISSUE(Wa1aT, wbuf1, 32768);
  TDM_WAIT1();                             // Wfin ready
  __syncthreads();

  // stage 4: m2 = silu(t1 @ Wfin + bfin) + m_input -> tileA   (B = wbuf0)
  wave_gemm<128, 8>(tileB, 128, wbuf0, row_base, lane, acc);
#pragma unroll
  for (int nt = 0; nt < 8; ++nt) {
    int n = nt * 16 + n15; float b = bfin[n];
#pragma unroll
    for (int v = 0; v < 8; ++v) {
      int m = row_base + m_off + v; int e = e0 + m;
      float skip = (e < E) ? m_input[(size_t)e * 128 + n] : 0.0f;
      tileA[m * 128 + n] = (__bf16)(silu_f(acc[nt][v] + b) + skip);
    }
  }
  __syncthreads();
  TDM_ISSUE(Wa1bT, wbuf0, 32768);
  TDM_WAIT1();                             // Wa1a ready
  __syncthreads();

  // stage 5: resa1 p1 -> tileB   (B = wbuf1)
  wave_gemm<128, 8>(tileA, 128, wbuf1, row_base, lane, acc);
#pragma unroll
  for (int nt = 0; nt < 8; ++nt) {
    int n = nt * 16 + n15; float b = ba1a[n];
#pragma unroll
    for (int v = 0; v < 8; ++v) {
      int m = row_base + m_off + v;
      tileB[m * 128 + n] = (__bf16)silu_f(acc[nt][v] + b);
    }
  }
  __syncthreads();
  TDM_ISSUE(Wa2aT, wbuf1, 32768);
  TDM_WAIT1();                             // Wa1b ready
  __syncthreads();

  // stage 6: resa1 p2: m3 = m2 + silu(h @ W2 + b2) -> tileA   (B = wbuf0)
  wave_gemm<128, 8>(tileB, 128, wbuf0, row_base, lane, acc);
#pragma unroll
  for (int nt = 0; nt < 8; ++nt) {
    int n = nt * 16 + n15; float b = ba1b[n];
#pragma unroll
    for (int v = 0; v < 8; ++v) {
      int m = row_base + m_off + v;
      tileA[m * 128 + n] = (__bf16)((float)tileA[m * 128 + n] + silu_f(acc[nt][v] + b));
    }
  }
  __syncthreads();
  TDM_ISSUE(Wa2bT, wbuf0, 32768);
  TDM_WAIT1();                             // Wa2a ready
  __syncthreads();

  // stage 7: resa2 p1 -> tileB   (B = wbuf1)
  wave_gemm<128, 8>(tileA, 128, wbuf1, row_base, lane, acc);
#pragma unroll
  for (int nt = 0; nt < 8; ++nt) {
    int n = nt * 16 + n15; float b = ba2a[n];
#pragma unroll
    for (int v = 0; v < 8; ++v) {
      int m = row_base + m_off + v;
      tileB[m * 128 + n] = (__bf16)silu_f(acc[nt][v] + b);
    }
  }
  __syncthreads();
  TDM_WAIT0();                             // Wa2b ready
  __syncthreads();

  // stage 8: out = m3 + silu(h @ W2 + b2)  (fp32 to global, B = wbuf0)
  wave_gemm<128, 8>(tileB, 128, wbuf0, row_base, lane, acc);
#pragma unroll
  for (int nt = 0; nt < 8; ++nt) {
    int n = nt * 16 + n15; float b = ba2b[n];
#pragma unroll
    for (int v = 0; v < 8; ++v) {
      int m = row_base + m_off + v; int e = e0 + m;
      if (e < E)
        out[(size_t)e * 128 + n] = (float)tileA[m * 128 + n] + silu_f(acc[nt][v] + b);
    }
  }
}

extern "C" void kernel_launch(void* const* d_in, const int* in_sizes, int n_in,
                              void* d_out, int out_size, void* d_ws, size_t ws_size,
                              hipStream_t stream) {
  (void)n_in; (void)out_size; (void)ws_size;
  const int D = 128, DA = 64;
  const int E = in_sizes[0] / D;
  const int T = in_sizes[27];

  const float* m_input = (const float*)d_in[0];
  const float* rbf     = (const float*)d_in[1];
  const float* sbf     = (const float*)d_in[2];
  const float* Wkj     = (const float*)d_in[3];
  const float* bkj     = (const float*)d_in[4];
  const float* Wrbf1   = (const float*)d_in[5];
  const float* Wrbf2   = (const float*)d_in[6];
  const float* Wdown   = (const float*)d_in[7];
  const float* Wsbf1   = (const float*)d_in[8];
  const float* Wsbf2   = (const float*)d_in[9];
  const float* Wup     = (const float*)d_in[10];
  const float* Wji     = (const float*)d_in[11];
  const float* bji     = (const float*)d_in[12];
  const float* Wb1a    = (const float*)d_in[13];
  const float* bb1a    = (const float*)d_in[14];
  const float* Wb1b    = (const float*)d_in[15];
  const float* bb1b    = (const float*)d_in[16];
  const float* Wfin    = (const float*)d_in[17];
  const float* bfin    = (const float*)d_in[18];
  const float* Wa1a    = (const float*)d_in[19];
  const float* ba1a    = (const float*)d_in[20];
  const float* Wa1b    = (const float*)d_in[21];
  const float* ba1b    = (const float*)d_in[22];
  const float* Wa2a    = (const float*)d_in[23];
  const float* ba2a    = (const float*)d_in[24];
  const float* Wa2b    = (const float*)d_in[25];
  const float* ba2b    = (const float*)d_in[26];
  const int* reduce_to_ji = (const int*)d_in[27];
  const int* expand_to_kj = (const int*)d_in[28];

  char* ws = (char*)d_ws;
  size_t off = 0;
  auto take = [&](size_t bytes) { char* p = ws + off; off += (bytes + 255) & ~(size_t)255; return p; };

  __bf16* WkjT  = (__bf16*)take((size_t)D * D * 2);
  __bf16* WjiT  = (__bf16*)take((size_t)D * D * 2);
  __bf16* Wb1aT = (__bf16*)take((size_t)D * D * 2);
  __bf16* Wb1bT = (__bf16*)take((size_t)D * D * 2);
  __bf16* WfinT = (__bf16*)take((size_t)D * D * 2);
  __bf16* Wa1aT = (__bf16*)take((size_t)D * D * 2);
  __bf16* Wa1bT = (__bf16*)take((size_t)D * D * 2);
  __bf16* Wa2aT = (__bf16*)take((size_t)D * D * 2);
  __bf16* Wa2bT = (__bf16*)take((size_t)D * D * 2);
  __bf16* WdownT = (__bf16*)take((size_t)D * DA * 2);   // [64][128]
  __bf16* WupT   = (__bf16*)take((size_t)DA * D * 2);   // [128][64]
  float* m_ang = (float*)take((size_t)E * DA * 4);
  float* agg   = (float*)take((size_t)E * DA * 4);

  (void)hipMemsetAsync(agg, 0, (size_t)E * DA * 4, stream);

  auto tr = [&](const float* W, __bf16* WT, int K, int N) {
    int total = K * N;
    k_w_to_bf16_t<<<(total + 255) / 256, 256, 0, stream>>>(W, WT, K, N);
  };
  tr(Wkj,  WkjT,  D, D);  tr(Wji,  WjiT,  D, D);
  tr(Wb1a, Wb1aT, D, D);  tr(Wb1b, Wb1bT, D, D);
  tr(Wfin, WfinT, D, D);
  tr(Wa1a, Wa1aT, D, D);  tr(Wa1b, Wa1bT, D, D);
  tr(Wa2a, Wa2aT, D, D);  tr(Wa2b, Wa2bT, D, D);
  tr(Wdown, WdownT, D, DA);
  tr(Wup,   WupT,   DA, D);

  const int eblocks = (E + 127) / 128;
  const int tblocks = (T + 7) / 8;

  k_edge_pre<<<eblocks, 256, 118784, stream>>>(
      m_input, rbf, Wrbf1, Wrbf2, bkj, WkjT, WdownT, m_ang, E);
  k_triplet<<<tblocks, 256, 0, stream>>>(
      sbf, Wsbf1, Wsbf2, m_ang, expand_to_kj, reduce_to_ji, agg, T);
  k_edge_post<<<eblocks, 256, 131072, stream>>>(
      m_input, agg, WupT, WjiT, bji, Wb1aT, bb1a, Wb1bT, bb1b,
      WfinT, bfin, Wa1aT, ba1a, Wa1bT, ba1b, Wa2aT, ba2a, Wa2bT, ba2b,
      (float*)d_out, E);
}
